// EPAN_8735963480604
// MI455X (gfx1250) — matrix operand order repaired and checked by reading, not compiled, not run on hardware
//
#include <hip/hip_runtime.h>
#include <math.h>

typedef __attribute__((ext_vector_type(16))) _Float16 v16h;
typedef __attribute__((ext_vector_type(8)))  _Float16 v8h;
typedef __attribute__((ext_vector_type(8)))  float    v8f;

#define B_  8
#define N_  4096
#define C_  512
#define H_  8
#define D_  64
#define P_  64
#define OC_ 1536   // 3*C

union HV16 { v16h v; v8h h[2]; };
union HV8  { v8h v; _Float16 e[8]; };

// A fragment (16x32 f16): lane holds row m, K runs [half*8, half*8+7] and [16+half*8, 23+half*8]
static __device__ __forceinline__ v16h ld_frag_a(const _Float16* row, int half) {
    HV16 u;
    u.h[0] = *(const v8h*)(row + half * 8);
    u.h[1] = *(const v8h*)(row + 16 + half * 8);
    return u.v;
}
// B fragment (32x16 f16): lane holds column n', K runs [half*8, half*8+7] and [16+half*8, 23+half*8]
static __device__ __forceinline__ v16h ld_frag_b(const _Float16* row, int half) {
    HV16 u;
    u.h[0] = *(const v8h*)(row + half * 8);
    u.h[1] = *(const v8h*)(row + 16 + half * 8);
    return u.v;
}
static __device__ __forceinline__ v8f wmma_f16(v16h a, v16h b, v8f c) {
    return __builtin_amdgcn_wmma_f32_16x16x32_f16(false, a, false, b, (short)0, c, false, false);
}

// ---------------- kernel 1: convert weights f32 -> f16 ----------------
__global__ __launch_bounds__(256) void k_convert(const float* __restrict__ wqkv,
                                                 const float* __restrict__ we,
                                                 _Float16* __restrict__ wq16,
                                                 _Float16* __restrict__ we16) {
    int i = blockIdx.x * 256 + threadIdx.x;
    if (i < OC_ * C_) wq16[i] = (_Float16)wqkv[i];
    int j = i - OC_ * C_;
    if (j >= 0 && j < P_ * N_) we16[j] = (_Float16)we[j];
}

// ---------------- kernel 2: QKV GEMM  [N x C] @ [C x 3C] ----------------
// block tile: 128 tokens x 64 oc; K-chunk 64; double-buffered LDS
__global__ __launch_bounds__(256) void k_qkv(const float* __restrict__ x,
                                             const _Float16* __restrict__ wq16,
                                             _Float16* __restrict__ qb,
                                             _Float16* __restrict__ kb,
                                             _Float16* __restrict__ vb) {
    __shared__ __align__(16) _Float16 sA[2][128][72];  // tokens x k(=c) contiguous
    __shared__ __align__(16) _Float16 sB[2][64][72];   // oc rows x k contiguous
    const int octile = blockIdx.x;            // 0..23
    const int n0 = blockIdx.y * 128;
    const int b  = blockIdx.z;
    const int t  = threadIdx.x;
    const int w  = t >> 5, lane = t & 31, lr = lane & 15, half = lane >> 4;
    const int oc0 = octile * 64;
    const float* xb = x + ((size_t)b * N_ + n0) * C_;

    auto stage = [&](int buf, int c0) {
        #pragma unroll
        for (int j = 0; j < 8; ++j) {                       // A: 128 x 64 f32 -> f16
            int idx = t * 8 + j;                            // 2048 float4
            int row = idx >> 4, c4 = (idx & 15) * 4;
            const float4 v = *(const float4*)(xb + (size_t)row * C_ + c0 + c4);
            sA[buf][row][c4 + 0] = (_Float16)v.x;
            sA[buf][row][c4 + 1] = (_Float16)v.y;
            sA[buf][row][c4 + 2] = (_Float16)v.z;
            sA[buf][row][c4 + 3] = (_Float16)v.w;
        }
        #pragma unroll
        for (int j = 0; j < 2; ++j) {                       // B: 64 oc-rows x 64 c
            int idx = t * 2 + j;
            int ocl = idx >> 3, c8 = (idx & 7) * 8;
            *(v8h*)&sB[buf][ocl][c8] = *(const v8h*)(wq16 + (size_t)(oc0 + ocl) * C_ + c0 + c8);
        }
    };

    v8f acc[4] = {};
    stage(0, 0);
    for (int ks = 0; ks < 8; ++ks) {
        __syncthreads();
        if (ks + 2 < 8)  // prefetch chunk ks+2 (one byte per 128B of the A tile)
            __builtin_prefetch(xb + (size_t)(t >> 1) * C_ + (ks + 2) * 64 + (t & 1) * 32, 0, 3);
        if (ks + 1 < 8) stage((ks + 1) & 1, (ks + 1) * 64);
        const int buf = ks & 1;
        #pragma unroll
        for (int ksub = 0; ksub < 2; ++ksub) {
            v16h af = ld_frag_a(&sA[buf][w * 16 + lr][ksub * 32], half);
            #pragma unroll
            for (int j = 0; j < 4; ++j) {
                v16h bf = ld_frag_b(&sB[buf][j * 16 + lr][ksub * 32], half);
                acc[j] = wmma_f16(af, bf, acc[j]);
            }
        }
    }
    const int tsel = octile >> 3;   // 0=q 1=k 2=v   (oc = tsel*512 + h*64 + dd)
    const int h = octile & 7;
    _Float16* dst = (tsel == 0) ? qb : (tsel == 1) ? kb : vb;
    const int bh = b * H_ + h;
    #pragma unroll
    for (int j = 0; j < 4; ++j)
        #pragma unroll
        for (int r = 0; r < 8; ++r) {
            int token = n0 + w * 16 + r + 8 * half;
            int dd = j * 16 + lr;
            dst[((size_t)bh * N_ + token) * D_ + dd] = (_Float16)acc[j][r];
        }
}

// ---------------- kernel 3: 1 / max(||q_row||, eps) over token axis ----------------
__global__ __launch_bounds__(256) void k_rnorm(const _Float16* __restrict__ qb,
                                               float* __restrict__ rn) {
    __shared__ float red[256];
    const int bh = blockIdx.x;
    const int t = threadIdx.x;
    const int dd = t & 63, seg = t >> 6;
    float s = 0.f;
    for (int n = seg * 1024; n < seg * 1024 + 1024; ++n) {
        float v = (float)qb[((size_t)bh * N_ + n) * D_ + dd];
        s += v * v;
    }
    red[t] = s;
    __syncthreads();
    if (t < 64) {
        float tot = red[t] + red[t + 64] + red[t + 128] + red[t + 192];
        rn[bh * D_ + dd] = 1.0f / fmaxf(sqrtf(tot), 1e-12f);
    }
}

// ---------------- kernel 4: k_proj / v_proj  (K = N = 4096) ----------------
// K-chunk 64, double-buffered. k_proj scaled by temp*rnorm, stored [p][dd]; v_proj stored [dd][p]
__global__ __launch_bounds__(256) void k_proj(const _Float16* __restrict__ kb,
                                              const _Float16* __restrict__ vb,
                                              const _Float16* __restrict__ we16,
                                              const float* __restrict__ be,
                                              const float* __restrict__ temp,
                                              const float* __restrict__ rn,
                                              _Float16* __restrict__ kpt,
                                              _Float16* __restrict__ vps) {
    __shared__ __align__(16) _Float16 sAt[2][64][72];  // dd rows x k(=n) contiguous
    __shared__ __align__(16) _Float16 sB[2][64][72];   // p rows x k contiguous
    const int which = blockIdx.x;                   // 0 -> k_proj, 1 -> v_proj
    const int h = blockIdx.y, b = blockIdx.z;
    const int bh = b * H_ + h;
    const int t = threadIdx.x, w = t >> 5, lane = t & 31, lr = lane & 15, half = lane >> 4;
    const _Float16* src = (which == 0) ? kb : vb;
    const int mt = w & 3;            // dd tile (4)
    const int pp = (w >> 2) * 2;     // p tile pair base
    const int nl = t >> 2, d16 = (t & 3) * 16;         // staging coords (reused below)

    auto stage = [&](int buf, int nk) {
        // A transposed: src[n][dd] -> sAt[dd][n_local], 64 x 64
        HV8 u0, u1;
        u0.v = *(const v8h*)(src + ((size_t)bh * N_ + nk + nl) * D_ + d16);
        u1.v = *(const v8h*)(src + ((size_t)bh * N_ + nk + nl) * D_ + d16 + 8);
        #pragma unroll
        for (int e = 0; e < 8; ++e) sAt[buf][d16 + e][nl] = u0.e[e];
        #pragma unroll
        for (int e = 0; e < 8; ++e) sAt[buf][d16 + 8 + e][nl] = u1.e[e];
        // B direct: w_e[p][n] is already k-contiguous, 64 x 64
        *(v8h*)&sB[buf][nl][d16]     = *(const v8h*)(we16 + (size_t)nl * N_ + nk + d16);
        *(v8h*)&sB[buf][nl][d16 + 8] = *(const v8h*)(we16 + (size_t)nl * N_ + nk + d16 + 8);
    };

    v8f acc[2] = {};
    stage(0, 0);
    for (int ks = 0; ks < 64; ++ks) {
        __syncthreads();
        if (ks + 2 < 64) {
            __builtin_prefetch(src + ((size_t)bh * N_ + (ks + 2) * 64 + nl) * D_ + d16, 0, 3);
            __builtin_prefetch(we16 + (size_t)nl * N_ + (ks + 2) * 64 + d16, 0, 3);
        }
        if (ks + 1 < 64) stage((ks + 1) & 1, (ks + 1) * 64);
        const int buf = ks & 1;
        #pragma unroll
        for (int ksub = 0; ksub < 2; ++ksub) {
            v16h af = ld_frag_a(&sAt[buf][mt * 16 + lr][ksub * 32], half);
            #pragma unroll
            for (int j = 0; j < 2; ++j) {
                v16h bf = ld_frag_b(&sB[buf][(pp + j) * 16 + lr][ksub * 32], half);
                acc[j] = wmma_f16(af, bf, acc[j]);
            }
        }
    }
    const float th = temp[h];
    #pragma unroll
    for (int j = 0; j < 2; ++j) {
        int p = (pp + j) * 16 + lr;
        float bias = be[p];
        #pragma unroll
        for (int r = 0; r < 8; ++r) {
            int dd = mt * 16 + r + 8 * half;
            float v = acc[j][r] + bias;
            if (which == 0) {
                v *= th * rn[bh * D_ + dd];
                kpt[((size_t)bh * P_ + p) * D_ + dd] = (_Float16)v;   // [p][dd]
            } else {
                vps[((size_t)bh * D_ + dd) * P_ + p] = (_Float16)v;   // [dd][p]
            }
        }
    }
}

// ---------------- kernel 5: fused scores -> softmax -> output ----------------
__global__ __launch_bounds__(256) void k_attn(const _Float16* __restrict__ qb,
                                              const _Float16* __restrict__ kpt,
                                              const _Float16* __restrict__ vps,
                                              float* __restrict__ out) {
    __shared__ __align__(16) _Float16 sQ[128][72];   // tokens x d (then reused for softmaxed A)
    __shared__ __align__(16) _Float16 sKt[64][72];   // p rows x dd contiguous
    __shared__ __align__(16) _Float16 sV[64][72];    // dd rows x p contiguous
    __shared__ __align__(16) float    sS[128 * 64];  // scores; reused as [64][128] O-transpose
    const int n0 = blockIdx.x * 128;
    const int h = blockIdx.y, b = blockIdx.z;
    const int bh = b * H_ + h;
    const int t = threadIdx.x, w = t >> 5, lane = t & 31, lr = lane & 15, half = lane >> 4;

    #pragma unroll
    for (int j = 0; j < 4; ++j) {          // stage q tile (128x64)
        int idx = t * 4 + j;
        int row = idx >> 3, c8 = (idx & 7) * 8;
        *(v8h*)&sQ[row][c8] = *(const v8h*)(qb + ((size_t)bh * N_ + n0 + row) * D_ + c8);
    }
    #pragma unroll
    for (int j = 0; j < 2; ++j) {          // stage scaled k_projT (64x64)
        int idx = t * 2 + j;
        int row = idx >> 3, c8 = (idx & 7) * 8;
        *(v8h*)&sKt[row][c8] = *(const v8h*)(kpt + ((size_t)bh * P_ + row) * D_ + c8);
    }
    #pragma unroll
    for (int j = 0; j < 2; ++j) {          // stage v_proj (64x64)
        int idx = t * 2 + j;
        int row = idx >> 3, c8 = (idx & 7) * 8;
        *(v8h*)&sV[row][c8] = *(const v8h*)(vps + ((size_t)bh * D_ + row) * P_ + c8);
    }
    __syncthreads();

    {   // scores: S[token][p] = q . k_scaled   (M=128, N'=64, K=64)
        v8f acc[4] = {};
        #pragma unroll
        for (int ks = 0; ks < 2; ++ks) {
            v16h af = ld_frag_a(&sQ[w * 16 + lr][ks * 32], half);
            #pragma unroll
            for (int j = 0; j < 4; ++j) {
                v16h bf = ld_frag_b(&sKt[j * 16 + lr][ks * 32], half);
                acc[j] = wmma_f16(af, bf, acc[j]);
            }
        }
        #pragma unroll
        for (int j = 0; j < 4; ++j)
            #pragma unroll
            for (int r = 0; r < 8; ++r)
                sS[(w * 16 + r + 8 * half) * 64 + j * 16 + lr] = acc[j][r];
    }
    __syncthreads();

    if (t < 128) {                          // softmax over P=64, write f16 A into sQ
        const float* rowp = &sS[t * 64];
        float m = rowp[0];
        for (int p = 1; p < 64; ++p) m = fmaxf(m, rowp[p]);
        float ssum = 0.f;
        for (int p = 0; p < 64; ++p) ssum += __expf(rowp[p] - m);
        float inv = 1.0f / ssum;
        for (int p = 0; p < 64; ++p) sQ[t][p] = (_Float16)(__expf(rowp[p] - m) * inv);
    }
    __syncthreads();

    {   // output: O[token][dd] = A . v_projT  (M=128, N'=64, K=64); dump transposed
        v8f acc[4] = {};
        #pragma unroll
        for (int ks = 0; ks < 2; ++ks) {
            v16h af = ld_frag_a(&sQ[w * 16 + lr][ks * 32], half);
            #pragma unroll
            for (int j = 0; j < 4; ++j) {
                v16h bf = ld_frag_b(&sV[j * 16 + lr][ks * 32], half);
                acc[j] = wmma_f16(af, bf, acc[j]);
            }
        }
        #pragma unroll
        for (int j = 0; j < 4; ++j)
            #pragma unroll
            for (int r = 0; r < 8; ++r)
                sS[(j * 16 + lr) * 128 + (w * 16 + r + 8 * half)] = acc[j][r];
    }
    __syncthreads();

    {   // coalesced write with torch-faithful permute: out[((b*64+dd)*8+h)*4096 + n]
        int dd = t >> 2, tc = (t & 3) * 32;
        float* gp = out + (((size_t)b * D_ + dd) * H_ + h) * N_ + n0 + tc;
        #pragma unroll
        for (int j = 0; j < 8; ++j)
            *(float4*)(gp + j * 4) = *(const float4*)&sS[dd * 128 + tc + j * 4];
    }
}

extern "C" void kernel_launch(void* const* d_in, const int* in_sizes, int n_in,
                              void* d_out, int out_size, void* d_ws, size_t ws_size,
                              hipStream_t stream) {
    const float* x    = (const float*)d_in[0];
    const float* wqkv = (const float*)d_in[1];
    const float* we   = (const float*)d_in[2];
    const float* be   = (const float*)d_in[3];
    const float* temp = (const float*)d_in[4];
    float* out = (float*)d_out;

    char* ws = (char*)d_ws;
    size_t o = 0;
    _Float16* wq16 = (_Float16*)(ws + o); o += (size_t)OC_ * C_ * 2;          // 1.5 MB
    _Float16* we16 = (_Float16*)(ws + o); o += (size_t)P_ * N_ * 2;           // 0.5 MB
    _Float16* qb   = (_Float16*)(ws + o); o += (size_t)B_ * H_ * N_ * D_ * 2; // 32 MB
    _Float16* kb   = (_Float16*)(ws + o); o += (size_t)B_ * H_ * N_ * D_ * 2; // 32 MB
    _Float16* vb   = (_Float16*)(ws + o); o += (size_t)B_ * H_ * N_ * D_ * 2; // 32 MB
    float*    rn   = (float*)(ws + o);    o += (size_t)B_ * H_ * D_ * 4;      // 16 KB
    _Float16* kpt  = (_Float16*)(ws + o); o += (size_t)B_ * H_ * P_ * D_ * 2; // 0.5 MB
    _Float16* vps  = (_Float16*)(ws + o); o += (size_t)B_ * H_ * D_ * P_ * 2; // 0.5 MB
    (void)ws_size; (void)in_sizes; (void)n_in; (void)out_size;

    k_convert<<<dim3((OC_ * C_ + P_ * N_) / 256), 256, 0, stream>>>(wqkv, we, wq16, we16);
    k_qkv   <<<dim3(24, 32, 8), 256, 0, stream>>>(x, wq16, qb, kb, vb);
    k_rnorm <<<dim3(64), 256, 0, stream>>>(qb, rn);
    k_proj  <<<dim3(2, 8, 8), 256, 0, stream>>>(kb, vb, we16, be, temp, rn, kpt, vps);
    k_attn  <<<dim3(32, 8, 8), 256, 0, stream>>>(qb, kpt, vps, out);
}
